// Discriminator_29437705846955
// MI455X (gfx1250) — matrix operand order repaired and checked by reading, not compiled, not run on hardware
//
#include <hip/hip_runtime.h>
#include <hip/hip_bf16.h>
#include <math.h>

// ---------------------------------------------------------------------------
// CDNA5 (gfx1250) implementation of the MPNN discriminator.
// All GEMMs (edge net, GRU, LSTM, classifier) run on v_wmma_f32_16x16x32_bf16,
// 16x64 output tile per wave32 (1 A fragment reused across 4 B fragments).
// bf16 operands / f32 accumulate. Scatter-mean via global_atomic_add_f32.
// ---------------------------------------------------------------------------

typedef __bf16 bf16;
typedef __attribute__((ext_vector_type(16))) __bf16 v16bf;
typedef __attribute__((ext_vector_type(8)))  float  v8f;
typedef __attribute__((ext_vector_type(4)))  unsigned int u32x4;

union FragB16 { v16bf v; u32x4 q[2]; };

#define CDIV(a, b) (((a) + (b) - 1) / (b))

__device__ __forceinline__ float sigmf(float x) { return 1.0f / (1.0f + __expf(-x)); }

// ------------------------------ utility ------------------------------------

__global__ void k_zero_f32(float* __restrict__ p, int n) {
    int i = blockIdx.x * blockDim.x + threadIdx.x;
    if (i < n) p[i] = 0.0f;
}

__global__ void k_f32_to_bf16(const float* __restrict__ x, bf16* __restrict__ y, int n) {
    int i = blockIdx.x * blockDim.x + threadIdx.x;
    if (i < n) y[i] = (bf16)x[i];
}

// W [K,N] f32  ->  Wt [N,K] bf16  (so WMMA B-columns are contiguous)
__global__ void k_transpose_to_bf16(const float* __restrict__ W, bf16* __restrict__ Wt,
                                    int K, int N) {
    int i = blockIdx.x * blockDim.x + threadIdx.x;
    if (i >= K * N) return;
    int k = i / N, n = i % N;
    Wt[(size_t)n * K + k] = (bf16)W[i];
}

// ------------------------------ batch-norm stats ---------------------------
// two-level column reduction: partial sums per row-chunk, atomics into sums[2C]

__global__ void k_colstats_f32(const float* __restrict__ X, float* __restrict__ sums,
                               int R, int C, int rowsPerChunk) {
    int c  = blockIdx.x * blockDim.x + threadIdx.x;
    if (c >= C) return;
    int r0 = blockIdx.y * rowsPerChunk;
    int r1 = min(R, r0 + rowsPerChunk);
    float s = 0.0f, s2 = 0.0f;
    for (int r = r0; r < r1; ++r) {
        float v = X[(size_t)r * C + c];
        s += v; s2 += v * v;
    }
    atomicAdd(&sums[c], s);
    atomicAdd(&sums[C + c], s2);
}

__global__ void k_colstats_bf16(const bf16* __restrict__ X, float* __restrict__ sums,
                                int R, int C, int rowsPerChunk) {
    int c  = blockIdx.x * blockDim.x + threadIdx.x;
    if (c >= C) return;
    int r0 = blockIdx.y * rowsPerChunk;
    int r1 = min(R, r0 + rowsPerChunk);
    float s = 0.0f, s2 = 0.0f;
    for (int r = r0; r < r1; ++r) {
        float v = (float)X[(size_t)r * C + c];
        s += v; s2 += v * v;
    }
    atomicAdd(&sums[c], s);
    atomicAdd(&sums[C + c], s2);
}

// fold mean/var/gamma/beta into per-column scale & shift
__global__ void k_bn_finalize(const float* __restrict__ sums,
                              const float* __restrict__ g, const float* __restrict__ b,
                              float* __restrict__ scale, float* __restrict__ shift,
                              int R, int C) {
    int c = blockIdx.x * blockDim.x + threadIdx.x;
    if (c >= C) return;
    float invR = 1.0f / (float)R;
    float mean = sums[c] * invR;
    float var  = sums[C + c] * invR - mean * mean;
    float sc   = g[c] * rsqrtf(var + 1e-5f);
    scale[c] = sc;
    shift[c] = b[c] - mean * sc;
}

// y = leaky(x*scale + shift)  -> bf16
__global__ void k_bn_apply_leaky_bf16(const float* __restrict__ X,
                                      const float* __restrict__ scale,
                                      const float* __restrict__ shift,
                                      bf16* __restrict__ Y, int R, int C, float slope) {
    size_t i = (size_t)blockIdx.x * blockDim.x + threadIdx.x;
    if (i >= (size_t)R * C) return;
    int c = (int)(i % C);
    float v = X[i] * scale[c] + shift[c];
    v = (v >= 0.0f) ? v : slope * v;
    Y[i] = (bf16)v;
}

// in-place BN on the big bf16 per-edge weight tensor
__global__ void k_bn_apply_bf16_inplace(bf16* __restrict__ X,
                                        const float* __restrict__ scale,
                                        const float* __restrict__ shift,
                                        int R, int C) {
    size_t i = (size_t)blockIdx.x * blockDim.x + threadIdx.x;
    if (i >= (size_t)R * C) return;
    int c = (int)(i % C);
    X[i] = (bf16)((float)X[i] * scale[c] + shift[c]);
}

// ------------------------------ embeddings ---------------------------------
// H[r,d] = bias[d] + sum_f (X[r,f]*scale[f]+shift[f]) * W[f,d]   (small F)

__global__ void k_embed(const float* __restrict__ X,
                        const float* __restrict__ scale, const float* __restrict__ shift,
                        const float* __restrict__ W, const float* __restrict__ bias,
                        float* __restrict__ H, bf16* __restrict__ Hb,
                        int R, int F, int Dout) {
    size_t i = (size_t)blockIdx.x * blockDim.x + threadIdx.x;
    if (i >= (size_t)R * Dout) return;
    int r = (int)(i / Dout), d = (int)(i % Dout);
    const float* xr = X + (size_t)r * F;
    float acc = bias[d];
    for (int f = 0; f < F; ++f)
        acc += (xr[f] * scale[f] + shift[f]) * W[(size_t)f * Dout + d];
    if (H)  H[i]  = acc;
    if (Hb) Hb[i] = (bf16)acc;
}

// ------------------------------ WMMA GEMM (16x64 tile / wave) --------------
// C[M,N] = A[M,K](bf16) @ Bt[N,K](bf16, pre-transposed) + bias
// one wave32 per 16x64 tile: 1 A fragment feeds 4 WMMAs per k-step.
// Requires K % 32 == 0, M % 16 == 0, N % 64 == 0 (true for all call sites).

template <typename OutT>
__global__ __launch_bounds__(32)
void k_wmma_gemm4(const bf16* __restrict__ A, const bf16* __restrict__ Bt,
                  const float* __restrict__ bias, OutT* __restrict__ C,
                  int M, int N, int K) {
    int tn = blockIdx.x;               // 64-wide column group
    int tm = blockIdx.y;               // 16-row band
    int lane = threadIdx.x;
    int hiA  = (lane & 16) ? 16 : 0;   // K sub-block offset for A fragment
    int hiB  = (lane & 16) ? 16 : 0;   // K sub-block offset for B fragment
    const bf16* pa = A  + (size_t)(tm * 16 + (lane & 15)) * K + hiA;
    const bf16* pb = Bt + (size_t)(tn * 64 + (lane & 15)) * K + hiB;
    const size_t bstep = (size_t)16 * K;   // next 16-column B sub-tile

    v8f acc0 = {}, acc1 = {}, acc2 = {}, acc3 = {};
    for (int k0 = 0; k0 < K; k0 += 32) {
        FragB16 a, b0, b1, b2, b3;
        a.q[0]  = *(const u32x4*)(pa + k0);
        a.q[1]  = *(const u32x4*)(pa + k0 + 8);
        b0.q[0] = *(const u32x4*)(pb + k0);
        b0.q[1] = *(const u32x4*)(pb + k0 + 8);
        b1.q[0] = *(const u32x4*)(pb + bstep + k0);
        b1.q[1] = *(const u32x4*)(pb + bstep + k0 + 8);
        b2.q[0] = *(const u32x4*)(pb + 2 * bstep + k0);
        b2.q[1] = *(const u32x4*)(pb + 2 * bstep + k0 + 8);
        b3.q[0] = *(const u32x4*)(pb + 3 * bstep + k0);
        b3.q[1] = *(const u32x4*)(pb + 3 * bstep + k0 + 8);
        acc0 = __builtin_amdgcn_wmma_f32_16x16x32_bf16(false, a.v, false, b0.v,
                                                       (short)0, acc0, false, false);
        acc1 = __builtin_amdgcn_wmma_f32_16x16x32_bf16(false, a.v, false, b1.v,
                                                       (short)0, acc1, false, false);
        acc2 = __builtin_amdgcn_wmma_f32_16x16x32_bf16(false, a.v, false, b2.v,
                                                       (short)0, acc2, false, false);
        acc3 = __builtin_amdgcn_wmma_f32_16x16x32_bf16(false, a.v, false, b3.v,
                                                       (short)0, acc3, false, false);
    }

    int rowbase = tm * 16 + ((lane & 16) ? 8 : 0);
    v8f accs[4] = {acc0, acc1, acc2, acc3};
#pragma unroll
    for (int j = 0; j < 4; ++j) {
        int col  = tn * 64 + j * 16 + (lane & 15);
        float bv = bias ? bias[col] : 0.0f;
#pragma unroll
        for (int d = 0; d < 8; ++d)
            C[(size_t)(rowbase + d) * N + col] = (OutT)(accs[j][d] + bv);
    }
}

// ------------------------------ graph ops ----------------------------------

__global__ void k_degree(const int* __restrict__ dst, float* __restrict__ deg, int E) {
    int e = blockIdx.x * blockDim.x + threadIdx.x;
    if (e < E) atomicAdd(&deg[dst[e]], 1.0f);
}

__global__ void k_invdeg(const float* __restrict__ deg, float* __restrict__ inv, int N) {
    int i = blockIdx.x * blockDim.x + threadIdx.x;
    if (i < N) inv[i] = (deg[i] > 0.0f) ? (1.0f / deg[i]) : 0.0f;
}

// one block (64 threads) per edge: msg = h[src] @ W_e, scatter-add into m[dst]
__global__ __launch_bounds__(64)
void k_message(const bf16* __restrict__ hb, const bf16* __restrict__ w,
               const int* __restrict__ src, const int* __restrict__ dst,
               float* __restrict__ m, int E) {
    __shared__ float hs[64];
    int e = blockIdx.x;
    int t = threadIdx.x;
    int s = src[e];
    hs[t] = (float)hb[(size_t)s * 64 + t];
    const bf16* we = w + (size_t)e * 4096;
    __builtin_prefetch(we + (size_t)t * 64, 0, 1);   // global_prefetch_b8 on row t
    __syncthreads();
    float acc = 0.0f;
#pragma unroll 8
    for (int i = 0; i < 64; ++i)
        acc += hs[i] * (float)we[(size_t)i * 64 + t];   // row reads are coalesced
    atomicAdd(&m[(size_t)dst[e] * 64 + t], acc);
}

// mb = bf16(m * inv_deg[node])
__global__ void k_scale_msg(const float* __restrict__ m, const float* __restrict__ inv,
                            bf16* __restrict__ mb, int N) {
    size_t i = (size_t)blockIdx.x * blockDim.x + threadIdx.x;
    if (i >= (size_t)N * 64) return;
    mb[i] = (bf16)(m[i] * inv[i >> 6]);
}

__global__ void k_gru_update(const float* __restrict__ gi, const float* __restrict__ gh,
                             float* __restrict__ h, bf16* __restrict__ hb, int N) {
    size_t i = (size_t)blockIdx.x * blockDim.x + threadIdx.x;
    if (i >= (size_t)N * 64) return;
    int n = (int)(i >> 6), d = (int)(i & 63);
    const float* a = gi + (size_t)n * 192;
    const float* b = gh + (size_t)n * 192;
    float r  = sigmf(a[d]       + b[d]);
    float z  = sigmf(a[64 + d]  + b[64 + d]);
    float nn = tanhf(a[128 + d] + r * b[128 + d]);
    float hv = (1.0f - z) * nn + z * h[i];
    h[i]  = hv;
    hb[i] = (bf16)hv;
}

__global__ void k_lstm_update(const float* __restrict__ gA, const float* __restrict__ gB,
                              float* __restrict__ h, float* __restrict__ c,
                              bf16* __restrict__ hb, int B) {
    size_t idx = (size_t)blockIdx.x * blockDim.x + threadIdx.x;
    if (idx >= (size_t)B * 64) return;
    int n = (int)(idx >> 6), d = (int)(idx & 63);
    const float* a = gA + (size_t)n * 256;
    const float* b = gB + (size_t)n * 256;
    float ig = sigmf(a[d]        + b[d]);
    float fg = sigmf(a[64 + d]   + b[64 + d]);
    float gg = tanhf(a[128 + d]  + b[128 + d]);
    float og = sigmf(a[192 + d]  + b[192 + d]);
    float cv = fg * c[idx] + ig * gg;
    float hv = og * tanhf(cv);
    c[idx]  = cv;
    h[idx]  = hv;
    hb[idx] = (bf16)hv;
}

// e[n] = dot(h[n], q[graph(n)])
__global__ void k_attn_e(const float* __restrict__ h, const float* __restrict__ q,
                         const int* __restrict__ node_graph, float* __restrict__ e, int N) {
    int n = blockIdx.x * blockDim.x + threadIdx.x;
    if (n >= N) return;
    const float* hn = h + (size_t)n * 64;
    const float* qg = q + (size_t)node_graph[n] * 64;
    float s = 0.0f;
#pragma unroll 8
    for (int i = 0; i < 64; ++i) s += hn[i] * qg[i];
    e[n] = s;
}

// segment softmax (25 consecutive nodes/graph) + weighted readout, build q_star
__global__ __launch_bounds__(64)
void k_attn_readout(const float* __restrict__ h, const float* __restrict__ q,
                    const float* __restrict__ e, float* __restrict__ q_star, int B) {
    __shared__ float alpha[25];
    int g = blockIdx.x;
    int t = threadIdx.x;
    int base = g * 25;
    if (t == 0) {
        float mx = -1e30f;
        for (int i = 0; i < 25; ++i) mx = fmaxf(mx, e[base + i]);
        float s = 0.0f;
        for (int i = 0; i < 25; ++i) { float x = __expf(e[base + i] - mx); alpha[i] = x; s += x; }
        float inv = 1.0f / s;
        for (int i = 0; i < 25; ++i) alpha[i] *= inv;
    }
    __syncthreads();
    float acc = 0.0f;
    for (int i = 0; i < 25; ++i) acc += alpha[i] * h[(size_t)(base + i) * 64 + t];
    q_star[(size_t)g * 128 + 64 + t] = acc;                    // readout half
    q_star[(size_t)g * 128 + t]      = q[(size_t)g * 64 + t];  // q half
}

// classifier tail: out = sigmoid(b2 + sum_d leaky(h1,0.1) * W2[d])
__global__ void k_head(const float* __restrict__ h1, const float* __restrict__ W2,
                       const float* __restrict__ b2, float* __restrict__ out, int B) {
    int g = blockIdx.x * blockDim.x + threadIdx.x;
    if (g >= B) return;
    float acc = b2[0];
#pragma unroll 8
    for (int d = 0; d < 64; ++d) {
        float v = h1[(size_t)g * 64 + d];
        v = (v >= 0.0f) ? v : 0.1f * v;
        acc += v * W2[d];
    }
    out[g] = sigmf(acc);
}

// ---------------------------------------------------------------------------

extern "C" void kernel_launch(void* const* d_in, const int* in_sizes, int n_in,
                              void* d_out, int out_size, void* d_ws, size_t ws_size,
                              hipStream_t stream) {
    (void)in_sizes; (void)n_in; (void)out_size; (void)ws_size;

    constexpr int N = 20000, E = 80000, B = 800, D = 64, DE = 128, GD = D * D; // 4096
    constexpr int L = 3, N_ITERS = 6, N_LSTM = 4;

    // ---- inputs ----
    const float* x_node    = (const float*)d_in[0];
    const float* x_edge    = (const float*)d_in[1];
    const float* bn_node_g = (const float*)d_in[2];
    const float* bn_node_b = (const float*)d_in[3];
    const float* node_W    = (const float*)d_in[4];
    const float* node_b    = (const float*)d_in[5];
    const float* bn_edge_g = (const float*)d_in[6];
    const float* bn_edge_b = (const float*)d_in[7];
    const float* edge_W    = (const float*)d_in[8];
    const float* edge_b    = (const float*)d_in[9];
    const float* en_W1     = (const float*)d_in[10];
    const float* en_b1     = (const float*)d_in[11];
    const float* en_bn1_g  = (const float*)d_in[12];
    const float* en_bn1_b  = (const float*)d_in[13];
    const float* en_W2     = (const float*)d_in[14];
    const float* en_b2     = (const float*)d_in[15];
    const float* en_bn2_g  = (const float*)d_in[16];
    const float* en_bn2_b  = (const float*)d_in[17];
    const float* gru_Wi    = (const float*)d_in[18];
    const float* gru_Wh    = (const float*)d_in[19];
    const float* gru_bi    = (const float*)d_in[20];
    const float* gru_bh    = (const float*)d_in[21];
    const float* lstm_Wi0  = (const float*)d_in[22];
    const float* lstm_Wh0  = (const float*)d_in[23];
    const float* lstm_b0   = (const float*)d_in[24];
    const float* lstm_Wi   = (const float*)d_in[25];
    const float* lstm_Wh   = (const float*)d_in[26];
    const float* lstm_b    = (const float*)d_in[27];
    const float* bn_out_g  = (const float*)d_in[28];
    const float* bn_out_b  = (const float*)d_in[29];
    const float* cls1_W    = (const float*)d_in[30];
    const float* cls1_b    = (const float*)d_in[31];
    const float* cls2_W    = (const float*)d_in[32];
    const float* cls2_b    = (const float*)d_in[33];
    const int*   src       = (const int*)d_in[34];
    const int*   dst       = (const int*)d_in[35];
    const int*   node_graph= (const int*)d_in[36];

    // ---- workspace carve-out ----
    size_t off = 0;
    auto alloc = [&](size_t bytes) -> void* {
        off = (off + 255) & ~(size_t)255;
        void* p = (char*)d_ws + off;
        off += bytes;
        return p;
    };
    bf16*  w_bf16   = (bf16*)alloc((size_t)E * GD * 2);       // 655 MB per-edge weights
    bf16*  heb      = (bf16*)alloc((size_t)E * DE * 2);       // edge embedding bf16
    float* t1       = (float*)alloc((size_t)E * DE * 4);      // edge-net hidden (pre-BN)
    bf16*  a2b      = (bf16*)alloc((size_t)E * DE * 2);       // edge-net hidden bf16
    float* h        = (float*)alloc((size_t)N * D * 4);       // node state f32
    bf16*  hb       = (bf16*)alloc((size_t)N * D * 2);        // node state bf16
    float* m        = (float*)alloc((size_t)N * D * 4);       // aggregated messages
    bf16*  mb       = (bf16*)alloc((size_t)N * D * 2);
    float* gi       = (float*)alloc((size_t)N * 192 * 4);
    float* gh       = (float*)alloc((size_t)N * 192 * 4);
    float* deg      = (float*)alloc((size_t)N * 4);
    float* invdeg   = (float*)alloc((size_t)N * 4);
    float* sums     = (float*)alloc((size_t)2 * GD * 4);
    float* bscale   = (float*)alloc((size_t)GD * 4);
    float* bshift   = (float*)alloc((size_t)GD * 4);
    bf16*  W1T      = (bf16*)alloc((size_t)DE * DE * 2);
    bf16*  W2T      = (bf16*)alloc((size_t)GD * DE * 2);
    bf16*  GWiT     = (bf16*)alloc((size_t)192 * D * 2);
    bf16*  GWhT     = (bf16*)alloc((size_t)192 * D * 2);
    bf16*  LWi0T    = (bf16*)alloc((size_t)256 * 128 * 2);
    bf16*  LWh0T    = (bf16*)alloc((size_t)256 * 64 * 2);
    bf16*  LWiT     = (bf16*)alloc((size_t)3 * 256 * 64 * 2);
    bf16*  LWhT     = (bf16*)alloc((size_t)3 * 256 * 64 * 2);
    bf16*  C1T      = (bf16*)alloc((size_t)64 * 128 * 2);
    float* q_star   = (float*)alloc((size_t)B * 128 * 4);
    bf16*  qsb      = (bf16*)alloc((size_t)B * 128 * 2);
    float* hsf      = (float*)alloc((size_t)N_LSTM * B * 64 * 4);
    float* csf      = (float*)alloc((size_t)N_LSTM * B * 64 * 4);
    bf16*  hsb      = (bf16*)alloc((size_t)N_LSTM * B * 64 * 2);
    float* gA       = (float*)alloc((size_t)B * 256 * 4);
    float* gB       = (float*)alloc((size_t)B * 256 * 4);
    float* e_buf    = (float*)alloc((size_t)N * 4);
    bf16*  qnb      = (bf16*)alloc((size_t)B * 128 * 2);
    float* h1       = (float*)alloc((size_t)B * 64 * 4);

    auto zero = [&](float* p, size_t n) {
        k_zero_f32<<<CDIV((int)n, 256), 256, 0, stream>>>(p, (int)n);
    };
    auto colstats = [&](const float* X, int R, int C) {
        zero(sums, 2 * (size_t)C);
        k_colstats_f32<<<dim3(CDIV(C, 256), CDIV(R, 1024)), 256, 0, stream>>>(X, sums, R, C, 1024);
    };
    auto finalize = [&](const float* g_, const float* b_, int R, int C) {
        k_bn_finalize<<<CDIV(C, 256), 256, 0, stream>>>(sums, g_, b_, bscale, bshift, R, C);
    };
    auto gemm_f32 = [&](const bf16* A, const bf16* Bt, const float* bias, float* C_,
                        int M, int Nc, int K) {
        k_wmma_gemm4<float><<<dim3(Nc / 64, M / 16), 32, 0, stream>>>(A, Bt, bias, C_, M, Nc, K);
    };

    // ---- embeddings (BN folded into scale/shift) ----
    colstats(x_node, N, 39);
    finalize(bn_node_g, bn_node_b, N, 39);
    k_embed<<<CDIV(N * D, 256), 256, 0, stream>>>(x_node, bscale, bshift, node_W, node_b,
                                                  h, hb, N, 39, D);
    colstats(x_edge, E, 10);
    finalize(bn_edge_g, bn_edge_b, E, 10);
    k_embed<<<CDIV(E * DE, 256), 256, 0, stream>>>(x_edge, bscale, bshift, edge_W, edge_b,
                                                   nullptr, heb, E, 10, DE);

    // ---- weight transposes -> bf16 [N,K] ----
    auto tr = [&](const float* W, bf16* Wt, int K, int Nc) {
        k_transpose_to_bf16<<<CDIV(K * Nc, 256), 256, 0, stream>>>(W, Wt, K, Nc);
    };
    tr(en_W1, W1T, DE, DE);
    tr(en_W2, W2T, DE, GD);
    tr(gru_Wi, GWiT, D, 192);
    tr(gru_Wh, GWhT, D, 192);
    tr(lstm_Wi0, LWi0T, 128, 256);
    tr(lstm_Wh0, LWh0T, 64, 256);
    for (int l = 0; l < 3; ++l) {
        tr(lstm_Wi + (size_t)l * 64 * 256, LWiT + (size_t)l * 256 * 64, 64, 256);
        tr(lstm_Wh + (size_t)l * 64 * 256, LWhT + (size_t)l * 256 * 64, 64, 256);
    }
    tr(cls1_W, C1T, 128, 64);

    // ---- edge network layer 1: [E,128]@[128,128] + BN + leaky(0.8) ----
    gemm_f32(heb, W1T, en_b1, t1, E, DE, DE);
    colstats(t1, E, DE);
    finalize(en_bn1_g, en_bn1_b, E, DE);
    k_bn_apply_leaky_bf16<<<CDIV(E * DE, 256), 256, 0, stream>>>(t1, bscale, bshift, a2b,
                                                                 E, DE, 0.8f);

    // ---- edge network layer 2 (big WMMA GEMM): [E,128]@[128,4096] -> bf16 w ----
    k_wmma_gemm4<bf16><<<dim3(GD / 64, E / 16), 32, 0, stream>>>(a2b, W2T, en_b2, w_bf16,
                                                                 E, GD, DE);
    zero(sums, 2 * (size_t)GD);
    k_colstats_bf16<<<dim3(CDIV(GD, 256), CDIV(E, 1024)), 256, 0, stream>>>(w_bf16, sums,
                                                                            E, GD, 1024);
    finalize(en_bn2_g, en_bn2_b, E, GD);
    {
        size_t total = (size_t)E * GD;
        k_bn_apply_bf16_inplace<<<(unsigned)CDIV(total, 256), 256, 0, stream>>>(w_bf16, bscale,
                                                                                bshift, E, GD);
    }

    // ---- degrees -> inverse ----
    zero(deg, N);
    k_degree<<<CDIV(E, 256), 256, 0, stream>>>(dst, deg, E);
    k_invdeg<<<CDIV(N, 256), 256, 0, stream>>>(deg, invdeg, N);

    // ---- MPNN: NNConv + GRU, 3 layers ----
    for (int it = 0; it < L; ++it) {
        zero(m, (size_t)N * D);
        k_message<<<E, 64, 0, stream>>>(hb, w_bf16, src, dst, m, E);
        k_scale_msg<<<CDIV(N * D, 256), 256, 0, stream>>>(m, invdeg, mb, N);
        gemm_f32(mb, GWiT, gru_bi, gi, N, 192, D);
        gemm_f32(hb, GWhT, gru_bh, gh, N, 192, D);
        k_gru_update<<<CDIV(N * D, 256), 256, 0, stream>>>(gi, gh, h, hb, N);
    }

    // ---- Set2Set: 6 iters, 4-layer LSTM, segment attention ----
    zero(q_star, (size_t)B * 128);
    zero(hsf, (size_t)N_LSTM * B * 64);
    zero(csf, (size_t)N_LSTM * B * 64);
    zero((float*)hsb, (size_t)N_LSTM * B * 64 / 2);   // bf16 zeros == f32 zeros bitwise

    for (int itr = 0; itr < N_ITERS; ++itr) {
        k_f32_to_bf16<<<CDIV(B * 128, 256), 256, 0, stream>>>(q_star, qsb, B * 128);
        for (int l = 0; l < N_LSTM; ++l) {
            float* hl  = hsf + (size_t)l * B * 64;
            float* cl  = csf + (size_t)l * B * 64;
            bf16*  hbl = hsb + (size_t)l * B * 64;
            if (l == 0) {
                gemm_f32(qsb, LWi0T, lstm_b0, gA, B, 256, 128);
                gemm_f32(hbl, LWh0T, nullptr, gB, B, 256, 64);
            } else {
                gemm_f32(hsb + (size_t)(l - 1) * B * 64,
                         LWiT + (size_t)(l - 1) * 256 * 64,
                         lstm_b + (size_t)(l - 1) * 256, gA, B, 256, 64);
                gemm_f32(hbl, LWhT + (size_t)(l - 1) * 256 * 64, nullptr, gB, B, 256, 64);
            }
            k_lstm_update<<<CDIV(B * 64, 256), 256, 0, stream>>>(gA, gB, hl, cl, hbl, B);
        }
        const float* q = hsf + (size_t)(N_LSTM - 1) * B * 64;   // top-layer h
        k_attn_e<<<CDIV(N, 256), 256, 0, stream>>>(h, q, node_graph, e_buf, N);
        k_attn_readout<<<B, 64, 0, stream>>>(h, q, e_buf, q_star, B);
    }

    // ---- classifier head ----
    colstats(q_star, B, 128);
    finalize(bn_out_g, bn_out_b, B, 128);
    k_bn_apply_leaky_bf16<<<CDIV(B * 128, 256), 256, 0, stream>>>(q_star, bscale, bshift,
                                                                  qnb, B, 128, 1.0f);
    gemm_f32(qnb, C1T, cls1_b, h1, B, 64, 128);
    k_head<<<CDIV(B, 128), 128, 0, stream>>>(h1, cls2_W, cls2_b, (float*)d_out, B);
}
